// SlinkyForcePredictor_18820546691818
// MI455X (gfx1250) — matrix operand order repaired and checked by reading, not compiled, not run on hardware
//
#include <hip/hip_runtime.h>
#include <hip/hip_bf16.h>
#include <math.h>

typedef __attribute__((ext_vector_type(16))) _Float16 v16h;
typedef __attribute__((ext_vector_type(8)))  _Float16 v8h;
typedef __attribute__((ext_vector_type(8)))  float    v8f;

// Activations stored as f16, rows padded to 64 so all WMMA A-fragment loads are maskless b128.
#define RS 64              // padded u-length per (n,m) row (halves)
#define NR 576             // 9*RS halves per node
#define HS 128             // padded hidden width for radial MLP

__device__ inline v8f wmma32(v16h a, v16h b, v8f c) {
  return __builtin_amdgcn_wmma_f32_16x16x32_f16(false, a, false, b, (short)0, c, false, false);
}

// ---------------- Wigner-3j sparse table (closed-form real-basis CG, unit Frobenius
// norm per path, sign per reference convention, 1/sqrt(NPATH_IN[l3]) folded in).
struct W3E { int p, i, j, k; float v; };
#define GP  0.12909944f
#define D2P 0.14907120f
#define BP  0.07453560f
#define Q1  0.04879500f
#define Q2  0.09759001f
#define WP  0.08451543f
constexpr W3E TBL[137] = {
  {0,0,0,0, 0.57735027f},
  {1,0,1,1,0.23570226f},{1,0,2,2,0.23570226f},{1,0,3,3,0.23570226f},
  {2,0,4,4,0.18257419f},{2,0,5,5,0.18257419f},{2,0,6,6,0.18257419f},{2,0,7,7,0.18257419f},{2,0,8,8,0.18257419f},
  {3,1,0,1,0.23570226f},{3,2,0,2,0.23570226f},{3,3,0,3,0.23570226f},
  {4,1,1,0,0.33333333f},{4,2,2,0,0.33333333f},{4,3,3,0,0.33333333f},
  {5,1,2,3, 0.16666667f},{5,1,3,2,-0.16666667f},{5,2,1,3,-0.16666667f},
  {5,2,3,1, 0.16666667f},{5,3,1,2, 0.16666667f},{5,3,2,1,-0.16666667f},
  {6,1,3,4,GP},{6,3,1,4,GP},{6,1,2,5,GP},{6,2,1,5,GP},{6,2,3,7,GP},{6,3,2,7,GP},
  {6,2,2,6,D2P},{6,1,1,6,-BP},{6,3,3,6,-BP},{6,3,3,8,GP},{6,1,1,8,-GP},
  {7,1,4,3,GP},{7,3,4,1,GP},{7,1,5,2,GP},{7,2,5,1,GP},{7,2,7,3,GP},{7,3,7,2,GP},
  {7,2,6,2,D2P},{7,1,6,1,-BP},{7,3,6,3,-BP},{7,3,8,3,GP},{7,1,8,1,-GP},
  {8,1,4,5,BP},{8,1,5,4,-BP},{8,3,4,7,-BP},{8,3,7,4,BP},{8,2,4,8,D2P},{8,2,8,4,-D2P},
  {8,3,5,6,-GP},{8,3,6,5,GP},{8,2,5,7,BP},{8,2,7,5,-BP},{8,3,5,8,-BP},{8,3,8,5,BP},
  {8,1,6,7,-GP},{8,1,7,6,GP},{8,1,7,8,-BP},{8,1,8,7,BP},
  {9,4,0,4,0.18257419f},{9,5,0,5,0.18257419f},{9,6,0,6,0.18257419f},{9,7,0,7,0.18257419f},{9,8,0,8,0.18257419f},
  {10,4,1,3,GP},{10,4,3,1,GP},{10,5,1,2,GP},{10,5,2,1,GP},{10,7,2,3,GP},{10,7,3,2,GP},
  {10,6,2,2,D2P},{10,6,1,1,-BP},{10,6,3,3,-BP},{10,8,3,3,GP},{10,8,1,1,-GP},
  {11,4,1,5,BP},{11,5,1,4,-BP},{11,4,3,7,-BP},{11,7,3,4,BP},{11,4,2,8,D2P},{11,8,2,4,-D2P},
  {11,5,3,6,-GP},{11,6,3,5,GP},{11,5,2,7,BP},{11,7,2,5,-BP},{11,5,3,8,-BP},{11,8,3,5,BP},
  {11,6,1,7,-GP},{11,7,1,6,GP},{11,7,1,8,-BP},{11,8,1,7,BP},
  {12,4,4,0,0.25819889f},{12,5,5,0,0.25819889f},{12,6,6,0,0.25819889f},{12,7,7,0,0.25819889f},{12,8,8,0,0.25819889f},
  {13,4,5,1,BP},{13,5,4,1,-BP},{13,4,7,3,-BP},{13,7,4,3,BP},{13,4,8,2,D2P},{13,8,4,2,-D2P},
  {13,5,6,3,-GP},{13,6,5,3,GP},{13,5,7,2,BP},{13,7,5,2,-BP},{13,5,8,3,-BP},{13,8,5,3,BP},
  {13,6,7,1,-GP},{13,7,6,1,GP},{13,7,8,1,-BP},{13,8,7,1,BP},
  {14,4,4,6,Q2},{14,4,6,4,Q2},{14,6,4,4,Q2},
  {14,5,5,6,-Q1},{14,5,6,5,-Q1},{14,6,5,5,-Q1},
  {14,7,7,6,-Q1},{14,7,6,7,-Q1},{14,6,7,7,-Q1},
  {14,6,6,6,-Q2},
  {14,6,8,8,Q2},{14,8,6,8,Q2},{14,8,8,6,Q2},
  {14,5,5,8,WP},{14,5,8,5,WP},{14,8,5,5,WP},
  {14,7,7,8,-WP},{14,7,8,7,-WP},{14,8,7,7,-WP},
  {14,4,5,7,-WP},{14,4,7,5,-WP},{14,5,4,7,-WP},{14,5,7,4,-WP},{14,7,4,5,-WP},{14,7,5,4,-WP}
};

// ---------------- edge prep: edge_vec -> SH(E,9), radial EMB(E,10)
__global__ void edge_prep_kernel(const float* __restrict__ pos,
                                 float* __restrict__ SH, float* __restrict__ EMB, int E) {
  int e = blockIdx.x * blockDim.x + threadIdx.x;
  if (e >= E) return;
  float ex = pos[e*6+1] - pos[e*6+0];
  float ey = pos[e*6+3] - pos[e*6+2];
  float ez = pos[e*6+5] - pos[e*6+4];
  float rn = sqrtf(ex*ex + ey*ey + ez*ez);
  float inv = 1.f / fmaxf(rn, 1e-12f);
  float x = ex*inv, y = ey*inv, z = ez*inv;
  const float s3 = 1.7320508f, s5 = 2.2360680f, s15 = 3.8729833f;
  SH[e*9+0] = 1.f;
  SH[e*9+1] = s3*x; SH[e*9+2] = s3*y; SH[e*9+3] = s3*z;
  SH[e*9+4] = s15*x*z; SH[e*9+5] = s15*x*y;
  SH[e*9+6] = s5*(y*y - 0.5f*(x*x + z*z));
  SH[e*9+7] = s15*y*z; SH[e*9+8] = 0.5f*s15*(z*z - x*x);
  const float step = 4.0f/11.0f;
  #pragma unroll
  for (int i = 0; i < 10; ++i) {
    float center = step * (float)(i+1);
    float diff = (rn - center) / step;
    float up = diff + 1.f, um = 1.f - diff;
    float fu = up > 0.f ? __expf(-1.f/up) : 0.f;
    float fm = um > 0.f ? __expf(-1.f/um) : 0.f;
    EMB[e*10+i] = 26.669224f * fu * fm;   // 1.14136*e^2*sqrt(10)
  }
}

// ---------------- x init (padded f16 layout (N,9,RS))
__global__ void init_x_kernel(const float* __restrict__ bar, const float* __restrict__ Wemb,
                              _Float16* __restrict__ X, int N) {
  int id = blockIdx.x * blockDim.x + threadIdx.x;
  if (id >= N * NR) return;
  int n = id / NR, rem = id % NR, m = rem / RS, u = rem % RS;
  X[id] = (_Float16)((m == 0 && u < 50) ? bar[n] * Wemb[u] : 0.f);
}

// ---------------- WMMA lmix over f16 activations (scales folded into LDS weights;
// B fragments pre-swizzled in LDS -> 2x ds_load_b128 per fragment;
// A fragments -> 2x global_load_b128 per fragment, no converts on load path).
__global__ __launch_bounds__(256) void lmix_wmma_kernel(
    const _Float16* __restrict__ Xa, const float* __restrict__ Wa, float sa,
    const _Float16* __restrict__ Xb, const float* __restrict__ Wb, float sb,
    _Float16* __restrict__ Out, int numNodes, int inMul, int dual)
{
  __shared__ __align__(32) _Float16 sW[2][3*4*2*512];   // [l][ct][kk][lane][16]
  int tid = threadIdx.x;
  int nsets = dual ? 2 : 1;
  for (int s = 0; s < nsets; ++s) {
    const float* Wsrc = s ? Wb : Wa;
    float sc = s ? sb : sa;
    for (int idx = tid; idx < 3*4*2*512; idx += blockDim.x) {
      int j  = idx & 15;
      int ln = (idx >> 4) & 31;
      int kk = (idx >> 9) & 1;
      int ct = (idx >> 10) & 3;
      int l  = idx >> 12;
      int u = kk*32 + (j & 7) + 8*(ln >> 4) + 16*(j >> 3);
      int v = ct*16 + (ln & 15);
      float val = (u < 50 && v < 50) ? Wsrc[(l*50 + u)*50 + v] * sc : 0.f;
      sW[s][idx] = (_Float16)val;
    }
  }
  __syncthreads();

  int lane = tid & 31, r = lane & 15, hi = lane >> 4;
  int wid = (blockIdx.x * blockDim.x + tid) >> 5;
  int nw  = (gridDim.x * blockDim.x) >> 5;

  int T0 = numNodes / 16, T1 = (numNodes*3) / 16, T2 = (numNodes*5) / 16;
  int totalTiles = (T0 + T1 + T2) * 4;

  for (int t = wid; t < totalTiles; t += nw) {
    int ct = t & 3;
    int rtg = t >> 2;
    int l, rt;
    if      (rtg < T0)      { l = 0; rt = rtg; }
    else if (rtg < T0 + T1) { l = 1; rt = rtg - T0; }
    else                    { l = 2; rt = rtg - T0 - T1; }
    int row0 = rt * 16;
    int vcol = ct * 16 + r;

    int rlA = row0 + r, nA, mA;
    if (l == 0)      { nA = rlA;     mA = 0; }
    else if (l == 1) { nA = rlA / 3; mA = 1 + (rlA - nA*3); }
    else             { nA = rlA / 5; mA = 4 + (rlA - nA*5); }
    int baseA = nA * inMul * NR + mA * RS;

    v8f acc = {};
    for (int s = 0; s < nsets; ++s) {
      const v8h* xp = (const v8h*)((s ? Xb : Xa) + baseA);
      const _Float16* wl = &sW[s][((l*4 + ct)*2) * 512];
      #pragma unroll
      for (int kk = 0; kk < 2; ++kk) {
        // A: K(j) = (j<8 ? j : j+8) + 8*hi  -> two contiguous 8-half runs
        v8h q0 = xp[kk*4 + hi];
        v8h q1 = xp[kk*4 + 2 + hi];
        v16h a;
        #pragma unroll
        for (int j = 0; j < 8; ++j) { a[j] = q0[j]; a[j+8] = q1[j]; }
        v16h b = *(const v16h*)(wl + kk*512 + lane*16);
        acc = wmma32(a, b, acc);
      }
    }
    // maskless stores (pad columns are exact zeros: zero weight columns)
    if (l == 0) {
      #pragma unroll
      for (int j = 0; j < 8; ++j) {
        int rl = row0 + j + 8*hi;
        Out[rl*NR + vcol] = (_Float16)acc[j];
      }
    } else if (l == 1) {
      #pragma unroll
      for (int j = 0; j < 8; ++j) {
        int rl = row0 + j + 8*hi; int nD = rl/3; int mD = 1 + (rl - nD*3);
        Out[nD*NR + mD*RS + vcol] = (_Float16)acc[j];
      }
    } else {
      #pragma unroll
      for (int j = 0; j < 8; ++j) {
        int rl = row0 + j + 8*hi; int nD = rl/5; int mD = 4 + (rl - nD*5);
        Out[nD*NR + mD*RS + vcol] = (_Float16)acc[j];
      }
    }
  }
}

// ---------------- radial MLP hidden: HH = gelu(EMB@W1/sqrt(10)+b1)  (E,HS) f16
__global__ void mlp1_kernel(const float* __restrict__ EMB, const float* __restrict__ W1,
                            const float* __restrict__ b1, _Float16* __restrict__ HH, int E) {
  int id = blockIdx.x * blockDim.x + threadIdx.x;
  if (id >= E * 100) return;
  int e = id / 100, o = id % 100;
  float acc = 0.f;
  #pragma unroll
  for (int i = 0; i < 10; ++i) acc += EMB[e*10+i] * W1[i*100+o];
  acc = acc * 0.31622777f + b1[o];
  float t = acc + 0.044715f * acc * acc * acc;
  float g = 0.5f * acc * (1.f + tanhf(0.7978845608f * t));
  HH[e*HS + o] = (_Float16)g;
}

// ---------------- pre-swizzle W2 (100x750) -> f16 fragment order [ct][kk][lane][16], *0.1
__global__ void w2_swizzle_kernel(const float* __restrict__ W2, _Float16* __restrict__ W2h) {
  int idx = blockIdx.x * blockDim.x + threadIdx.x;
  if (idx >= 47 * 2048) return;
  int j  = idx & 15;
  int ln = (idx >> 4) & 31;
  int kk = (idx >> 9) & 3;
  int ct = idx >> 11;
  int u = kk*32 + (j & 7) + 8*(ln >> 4) + 16*(j >> 3);
  int v = ct*16 + (ln & 15);
  float val = (u < 100 && v < 750) ? W2[u*750 + v] * 0.1f : 0.f;
  W2h[idx] = (_Float16)val;
}

// ---------------- WMMA edge-weight GEMM: WB(E,750 f16) = HH(E,100) @ W2h
__global__ __launch_bounds__(256) void wgemm_kernel(
    const _Float16* __restrict__ HH, const _Float16* __restrict__ W2h,
    _Float16* __restrict__ WBo, int E)
{
  int tid = threadIdx.x;
  int lane = tid & 31, r = lane & 15, hi = lane >> 4;
  int wid = (blockIdx.x * blockDim.x + tid) >> 5;
  int nw  = (gridDim.x * blockDim.x) >> 5;
  int rowTiles = E / 16;
  int total = rowTiles * 47;
  for (int t = wid; t < total; t += nw) {
    int ct = t % 47, rt = t / 47;
    int row0 = rt * 16;
    int eA = row0 + r;
    int vcol = ct * 16 + r;
    const _Float16* hp = HH + eA * HS;
    v8f acc = {};
    #pragma unroll
    for (int kk = 0; kk < 4; ++kk) {
      v8h q0 = *(const v8h*)(hp + kk*32 + 8*hi);
      v8h q1 = *(const v8h*)(hp + kk*32 + 16 + 8*hi);
      v16h a;
      #pragma unroll
      for (int j = 0; j < 8; ++j) { a[j] = q0[j]; a[j+8] = q1[j]; }
      v16h b = *(const v16h*)(W2h + ct*2048 + kk*512 + lane*16);
      acc = wmma32(a, b, acc);
    }
    if (vcol < 750) {
      #pragma unroll
      for (int j = 0; j < 8; ++j) {
        int er = row0 + j + 8*hi;
        WBo[er*750 + vcol] = (_Float16)acc[j];
      }
    }
  }
}

// ---------------- tensor-product message + scatter (dst = 2e+1 unique)
__global__ void message_kernel(const _Float16* __restrict__ H, const float* __restrict__ SH,
                               const _Float16* __restrict__ WBi, _Float16* __restrict__ AGG, int E) {
  int id = blockIdx.x * blockDim.x + threadIdx.x;
  if (id >= E * 50) return;
  int e = id / 50, c = id % 50;
  float h[9], shv[9], msg[9], wp[15];
  #pragma unroll
  for (int m = 0; m < 9; ++m) { h[m] = (float)H[e*NR + m*RS + c]; shv[m] = SH[e*9+m]; msg[m] = 0.f; }
  #pragma unroll
  for (int p = 0; p < 15; ++p) wp[p] = (float)WBi[e*750 + c*15 + p];
  #pragma unroll
  for (int q = 0; q < 137; ++q)
    msg[TBL[q].k] += wp[TBL[q].p] * TBL[q].v * h[TBL[q].i] * shv[TBL[q].j];
  const float invs2 = 0.70710678f;
  int dstBase = (2*e+1)*NR + c;
  int srcBase = (2*e)*NR + c;
  #pragma unroll
  for (int m = 0; m < 9; ++m) {
    AGG[dstBase + m*RS] = (_Float16)(msg[m] * invs2);
    AGG[srcBase + m*RS] = (_Float16)0.f;
  }
}

// ---------------- gate (in place): 4 nodes per 256-thread block
__global__ __launch_bounds__(256) void gate_kernel(_Float16* __restrict__ X,
                                                   const float* __restrict__ Wg, int N) {
  __shared__ float s[4][50];
  int slot = threadIdx.x >> 6;
  int v = threadIdx.x & 63;
  int n = blockIdx.x * 4 + slot;
  if (n < N && v < 50) s[slot][v] = (float)X[n*NR + v];
  __syncthreads();
  if (n >= N || v >= 50) return;
  float acc = 0.f;
  #pragma unroll
  for (int u = 0; u < 50; ++u) acc += s[slot][u] * Wg[u*50 + v];
  acc *= 0.14142136f;
  float g = 1.f / (1.f + __expf(-acc));
  float sv = s[slot][v];
  X[n*NR + v] = (_Float16)(sv / (1.f + __expf(-sv)));
  #pragma unroll
  for (int m = 1; m < 9; ++m) {
    float xv = (float)X[n*NR + m*RS + v];
    X[n*NR + m*RS + v] = (_Float16)(xv * g);
  }
}

// ---------------- final projection
__global__ void final_kernel(const _Float16* __restrict__ X, const float* __restrict__ Wout,
                             float* __restrict__ out, int N) {
  int id = blockIdx.x * blockDim.x + threadIdx.x;
  if (id >= N * 3) return;
  int n = id / 3, j = id % 3;
  float acc = 0.f;
  #pragma unroll
  for (int u = 0; u < 50; ++u) acc += (float)X[n*NR + (1+j)*RS + u] * Wout[u];
  out[id] = acc * 0.14142136f;
}

extern "C" void kernel_launch(void* const* d_in, const int* in_sizes, int n_in,
                              void* d_out, int out_size, void* d_ws, size_t ws_size,
                              hipStream_t stream) {
  const float* node_pos = (const float*)d_in[0];
  const float* bar      = (const float*)d_in[1];
  const float* W_embed  = (const float*)d_in[2];
  const float* Wsc      = (const float*)d_in[3];
  const float* Wlin1    = (const float*)d_in[4];
  const float* W1       = (const float*)d_in[5];
  const float* b1       = (const float*)d_in[6];
  const float* W2       = (const float*)d_in[7];
  const float* Wlin2    = (const float*)d_in[8];
  const float* Wgate    = (const float*)d_in[9];
  const float* W_out    = (const float*)d_in[10];
  float* out = (float*)d_out;

  const int E = in_sizes[0] / 6;     // 40000
  const int N = in_sizes[1];         // 80000
  const int Nh = N / 2;

  char* base = (char*)d_ws;
  size_t off = 0;
  auto alloc = [&](size_t bytes) { void* p = base + off; off += (bytes + 255) & ~(size_t)255; return p; };
  _Float16* X0   = (_Float16*)alloc((size_t)N * NR * 2);
  _Float16* X1   = (_Float16*)alloc((size_t)N * NR * 2);
  _Float16* AGG  = (_Float16*)alloc((size_t)N * NR * 2);
  _Float16* Hbuf = (_Float16*)alloc((size_t)E * NR * 2);
  _Float16* WB   = (_Float16*)alloc((size_t)E * 750 * 2);
  _Float16* HH   = (_Float16*)alloc((size_t)E * HS * 2);
  _Float16* W2h  = (_Float16*)alloc((size_t)47 * 2048 * 2);
  float*    SH   = (float*)alloc((size_t)E * 9 * 4);
  float*    EMB  = (float*)alloc((size_t)E * 10 * 4);
  (void)ws_size;

  const float inv_s50 = 0.14142136f;
  const float s1 = 0.92387953f * inv_s50;   // cos(pi/8)/sqrt(50)
  const float s2 = 0.38268343f * inv_s50;   // sin(pi/8)/sqrt(50)

  edge_prep_kernel<<<(E + 255) / 256, 256, 0, stream>>>(node_pos, SH, EMB, E);
  init_x_kernel<<<(N * NR + 255) / 256, 256, 0, stream>>>(bar, W_embed, X0, N);

  const int GB = 1920;
  _Float16* Xcur = X0;
  _Float16* Xnxt = X1;
  for (int i = 0; i < 3; ++i) {
    lmix_wmma_kernel<<<GB, 256, 0, stream>>>(Xcur, Wlin1 + i*7500, inv_s50,
                                             Xcur, Wlin1 + i*7500, 0.f,
                                             Hbuf, Nh, 2, 0);
    mlp1_kernel<<<(E * 100 + 255) / 256, 256, 0, stream>>>(EMB, W1 + i*1000, b1 + i*100, HH, E);
    w2_swizzle_kernel<<<(47 * 2048 + 255) / 256, 256, 0, stream>>>(W2 + i*75000, W2h);
    wgemm_kernel<<<GB, 256, 0, stream>>>(HH, W2h, WB, E);
    message_kernel<<<(E * 50 + 255) / 256, 256, 0, stream>>>(Hbuf, SH, WB, AGG, E);
    lmix_wmma_kernel<<<GB, 256, 0, stream>>>(Xcur, Wsc + i*7500, s1,
                                             AGG,  Wlin2 + i*7500, s2,
                                             Xnxt, N, 1, 1);
    gate_kernel<<<(N + 3) / 4, 256, 0, stream>>>(Xnxt, Wgate + i*2500, N);
    _Float16* tmp = Xcur; Xcur = Xnxt; Xnxt = tmp;
  }
  final_kernel<<<(N * 3 + 255) / 256, 256, 0, stream>>>(Xcur, W_out, out, N);
}
